// network_49581102465567
// MI455X (gfx1250) — matrix operand order repaired and checked by reading, not compiled, not run on hardware
//
#include <hip/hip_runtime.h>

typedef __attribute__((ext_vector_type(16))) _Float16 v16h;
typedef __attribute__((ext_vector_type(8)))  _Float16 v8h;
typedef __attribute__((ext_vector_type(8)))  float    v8f;
typedef __attribute__((ext_vector_type(4)))  float    v4f;
typedef __attribute__((ext_vector_type(4)))  unsigned int v4u;
typedef __attribute__((ext_vector_type(8)))  int      v8i;
typedef __attribute__((ext_vector_type(4)))  int      v4i;

#if defined(__gfx1250__) && __has_builtin(__builtin_amdgcn_tensor_load_to_lds) && \
    __has_builtin(__builtin_amdgcn_s_wait_tensorcnt)
#define USE_TDM 1
#else
#define USE_TDM 0
#endif

// ---------------------------------------------------------------------------
// f32 -> f16 conversion (8 elements/thread; all sizes are multiples of 8)
// ---------------------------------------------------------------------------
__global__ void cvt_f32_f16_kernel(const float* __restrict__ in,
                                   _Float16* __restrict__ out, long n) {
    long i = ((long)blockIdx.x * blockDim.x + threadIdx.x) * 8;
    if (i + 8 <= n) {
        v4f a = *(const v4f*)(in + i);
        v4f b = *(const v4f*)(in + i + 4);
        v8h o;
        o[0] = (_Float16)a[0]; o[1] = (_Float16)a[1];
        o[2] = (_Float16)a[2]; o[3] = (_Float16)a[3];
        o[4] = (_Float16)b[0]; o[5] = (_Float16)b[1];
        o[6] = (_Float16)b[2]; o[7] = (_Float16)b[3];
        *(v8h*)(out + i) = o;
    } else if (i < n) {
        for (long j = i; j < n; ++j) out[j] = (_Float16)in[j];
    }
}

// ---------------------------------------------------------------------------
// Nearest codeword from sorted 256-entry codebook (tie -> lower index).
// ---------------------------------------------------------------------------
__device__ __forceinline__ float quantize_cb(float x, const float* __restrict__ cb) {
    int lo = -1;  // largest idx with cb[idx] < x, or -1
#pragma unroll
    for (int s = 128; s > 0; s >>= 1) {
        int m = lo + s;
        if (m < 256 && cb[m] < x) lo = m;
    }
    int hi = lo + 1;
    if (lo < 0)   return cb[0];
    if (hi > 255) return cb[255];
    float cl = cb[lo], ch = cb[hi];
    return ((x - cl) <= (ch - x)) ? cl : ch;
}

// ---------------------------------------------------------------------------
// LDS B-tile geometry: 128 rows x 32 halfs, row stride 40 halfs (64B + 16B pad)
// ---------------------------------------------------------------------------
#define LDSB_STRIDE   40
#define LDSB_BYTES    (128 * LDSB_STRIDE * 2)   // 10240 B per buffer
#define LDS_TOTAL     (2 * LDSB_BYTES + 256 * 4)

#if USE_TDM
// One wave issues a TDM descriptor that DMAs a [128 x 32] f16 tile (row stride
// K halfs in memory) into LDS at byte offset ldsOff, padding 16B per 64B row.
// 6-arg builtin: (v4u group0, v8i group1, v4i group2, v4i group3, v8i ext, i32 cpol)
__device__ __forceinline__ void tdm_stage_tile(const _Float16* gsrc,
                                               unsigned ldsOff, int K, int N) {
    unsigned long long ga = (unsigned long long)(const void*)gsrc;
    v4u g0;
    g0[0] = 1u;                                   // count=1, no gather
    g0[1] = ldsOff;                               // lds_addr
    g0[2] = (unsigned)(ga & 0xFFFFFFFFull);       // global_addr[31:0]
    g0[3] = (unsigned)((ga >> 32) & 0x1FFFFFFull) // global_addr[56:32]
          | (2u << 30);                           // type = 2 ("image")
    v8i g1;
    g1[0] = (1 << 16)                             // data_size = 2 bytes
          | (1 << 20)                             // pad_enable
          | (3 << 22)                             // pad_interval: 16 DWORDs
          | (3 << 25);                            // pad_amount:   4 DWORDs
    g1[1] = (int)(((unsigned)K & 0xFFFFu) << 16); // tensor_dim0[15:0]
    g1[2] = (int)((((unsigned)K >> 16) & 0xFFFFu)           // tensor_dim0[31:16]
          | (((unsigned)N & 0xFFFFu) << 16));               // tensor_dim1[15:0]
    g1[3] = (int)((((unsigned)N >> 16) & 0xFFFFu)           // tensor_dim1[31:16]
          | (32u << 16));                                   // tile_dim0 = 32
    g1[4] = 128;                                  // tile_dim1 = 128, tile_dim2 = 0
    g1[5] = K;                                    // tensor_dim0_stride[31:0]
    g1[6] = 0;                                    // stride hi / dim1_stride lo
    g1[7] = 0;
    v4i z4 = {0, 0, 0, 0};
    v8i z8 = {0, 0, 0, 0, 0, 0, 0, 0};
    __builtin_amdgcn_tensor_load_to_lds(g0, g1, z4, z4, z8, 0);
}
#endif

// ---------------------------------------------------------------------------
// C[M,N] = A[M,K] @ W[N,K]^T + bias, optional quantize, f16 or f32 output.
// 256 threads = 8 waves. Block tile 256x128, K step 32.
// Wave w computes rows [32w, 32w+32) x 128 cols: 16 accumulators.
// B tile double-buffered in LDS (TDM-staged); A double-buffered in registers;
// all 16 ds_loads of a K-step issued before the WMMA chain so the in-order
// DS returns allow partial s_wait_dscnt overlap.
// ---------------------------------------------------------------------------
template <int QUANT, int OUT_F16>
__global__ __launch_bounds__(256) void gemm_f16_kernel(
    const _Float16* __restrict__ A,     // [M, K]
    const _Float16* __restrict__ W,     // [N, K]
    const float* __restrict__ bias,     // [N]
    const float* __restrict__ codebook, // [256] (QUANT only)
    _Float16* __restrict__ outH,        // [M, N] if OUT_F16
    float* __restrict__ outF,           // [M, N] if !OUT_F16
    int M, int N, int K) {

    extern __shared__ char smem[];                 // dynamic LDS -> offset 0
    _Float16* ldsB  = (_Float16*)smem;             // 2 buffers of 128x40 halfs
    float*    ldsCb = (float*)(smem + 2 * LDSB_BYTES);

    const int tid  = threadIdx.x;
    const int wave = tid >> 5;
    const int lane = tid & 31;
    const int lm   = lane & 15;
    const int lh   = lane >> 4;

    const int rowBase = blockIdx.y * 256 + wave * 32;
    const int colBase = blockIdx.x * 128;

    if (QUANT) ldsCb[tid & 255] = codebook[tid & 255];

    v8f acc[2][8];
#pragma unroll
    for (int t = 0; t < 2; ++t)
#pragma unroll
        for (int j = 0; j < 8; ++j) acc[t][j] = (v8f){};

    const _Float16* aP0 = A + (long)(rowBase + lm) * K + lh * 8;       // +k, +k+16
    const _Float16* aP1 = A + (long)(rowBase + 16 + lm) * K + lh * 8;

#if !USE_TDM
    const int wr = tid >> 1;                       // staging row 0..127
    const int wp = tid & 1;                        // 16-half chunk
    const _Float16* wRow = W + (long)(colBase + wr) * K + wp * 16;
    _Float16* ldsDst0 = ldsB + wr * LDSB_STRIDE + wp * 16;
#endif
    const _Float16* wTile = W + (long)colBase * K;

    // ---- prologue: stage B tile k=0 into buffer 0; load A for k=0 ----
#if USE_TDM
    if (wave == 0) tdm_stage_tile(wTile, 0u, K, N);
#else
    {
        v8h wv0 = *(const v8h*)(wRow);
        v8h wv1 = *(const v8h*)(wRow + 8);
        *(v8h*)(ldsDst0)     = wv0;
        *(v8h*)(ldsDst0 + 8) = wv1;
    }
#endif
    v8h a0lo = *(const v8h*)(aP0);
    v8h a0hi = *(const v8h*)(aP0 + 16);
    v8h a1lo = *(const v8h*)(aP1);
    v8h a1hi = *(const v8h*)(aP1 + 16);

    int cur = 0;
    for (int k = 0; k < K; k += 32) {
#if USE_TDM
        __builtin_amdgcn_s_wait_tensorcnt(0);  // no-op for non-issuing waves
#endif
        __syncthreads();   // tile k visible; all waves done with buf cur^1

        const bool more = (k + 32) < K;
        v8h n0lo = a0lo, n0hi = a0hi, n1lo = a1lo, n1hi = a1hi;
        if (more) {
            // stage next B tile + prefetch next A (full compute body to land)
#if USE_TDM
            if (wave == 0)
                tdm_stage_tile(wTile + (k + 32),
                               (unsigned)((cur ^ 1) * LDSB_BYTES), K, N);
#else
            v8h wv0 = *(const v8h*)(wRow + k + 32);
            v8h wv1 = *(const v8h*)(wRow + k + 40);
            _Float16* d = ldsDst0 + (cur ^ 1) * (LDSB_BYTES / 2);
            *(v8h*)(d)     = wv0;
            *(v8h*)(d + 8) = wv1;
#endif
            n0lo = *(const v8h*)(aP0 + k + 32);
            n0hi = *(const v8h*)(aP0 + k + 48);
            n1lo = *(const v8h*)(aP1 + k + 32);
            n1hi = *(const v8h*)(aP1 + k + 48);
        }

        v16h a0 = __builtin_shufflevector(a0lo, a0hi,
            0, 1, 2, 3, 4, 5, 6, 7, 8, 9, 10, 11, 12, 13, 14, 15);
        v16h a1 = __builtin_shufflevector(a1lo, a1hi,
            0, 1, 2, 3, 4, 5, 6, 7, 8, 9, 10, 11, 12, 13, 14, 15);

        const _Float16* bbase = ldsB + cur * (LDSB_BYTES / 2)
                              + lm * LDSB_STRIDE + lh * 8;

        // Issue ALL 16 ds_load_b128s first (in-order DS returns -> the first
        // WMMA pair only needs a partial dscnt wait; later fragments stream in
        // underneath the WMMA chain).
        v8h bf[8][2];
#pragma unroll
        for (int j = 0; j < 8; ++j) {
            const _Float16* p = bbase + j * 16 * LDSB_STRIDE;
            bf[j][0] = *(const v8h*)(p);
            bf[j][1] = *(const v8h*)(p + 16);
        }
#pragma unroll
        for (int j = 0; j < 8; ++j) {
            v16h b = __builtin_shufflevector(bf[j][0], bf[j][1],
                0, 1, 2, 3, 4, 5, 6, 7, 8, 9, 10, 11, 12, 13, 14, 15);
            acc[0][j] = __builtin_amdgcn_wmma_f32_16x16x32_f16(
                false, a0, false, b, (short)0, acc[0][j], false, false);
            acc[1][j] = __builtin_amdgcn_wmma_f32_16x16x32_f16(
                false, a1, false, b, (short)0, acc[1][j], false, false);
        }

        a0lo = n0lo; a0hi = n0hi; a1lo = n1lo; a1hi = n1hi;
        cur ^= 1;
    }

    // Epilogue: acc[t][j][v] = C[rowBase + t*16 + lh*8 + v][colBase + j*16 + lm]
#pragma unroll
    for (int t = 0; t < 2; ++t) {
#pragma unroll
        for (int j = 0; j < 8; ++j) {
            const int col = colBase + j * 16 + lm;
            const float bv = bias[col];
#pragma unroll
            for (int v = 0; v < 8; ++v) {
                const int row = rowBase + t * 16 + lh * 8 + v;
                float val = acc[t][j][v] + bv;
                if (QUANT) val = quantize_cb(val, ldsCb);
                if (OUT_F16) outH[(long)row * N + col] = (_Float16)val;
                else         outF[(long)row * N + col] = val;
            }
        }
    }
}

// ---------------------------------------------------------------------------
// Host launcher
// ---------------------------------------------------------------------------
extern "C" void kernel_launch(void* const* d_in, const int* in_sizes, int n_in,
                              void* d_out, int out_size, void* d_ws, size_t ws_size,
                              hipStream_t stream) {
    constexpr long Bsz = 16384, DIN = 1024, DOUT = 512;
    constexpr long H1 = 4096, H3 = 2048, H4 = 2048;

    const float* x  = (const float*)d_in[0];
    const float* cb = (const float*)d_in[1];
    const float* w1 = (const float*)d_in[2];
    const float* b1 = (const float*)d_in[3];
    const float* w2 = (const float*)d_in[4];
    const float* b2 = (const float*)d_in[5];
    const float* w3 = (const float*)d_in[6];
    const float* b3 = (const float*)d_in[7];
    const float* w4 = (const float*)d_in[8];
    const float* b4 = (const float*)d_in[9];
    const float* w5 = (const float*)d_in[10];
    const float* b5 = (const float*)d_in[11];
    float* out = (float*)d_out;

    char* ws = (char*)d_ws;
    size_t off = 0;
    auto take = [&](long elems) {
        _Float16* p = (_Float16*)(ws + off);
        off += (size_t)elems * sizeof(_Float16);
        off = (off + 255) & ~(size_t)255;
        return p;
    };
    _Float16* xh  = take(Bsz * DIN);
    _Float16* w1h = take(H1 * DIN);
    _Float16* w2h = take(DOUT * H1);
    _Float16* w3h = take(H3 * DOUT);
    _Float16* w4h = take(H4 * H3);
    _Float16* w5h = take(DOUT * H4);
    _Float16* h1  = take(Bsz * H1);
    _Float16* h2  = take(Bsz * DOUT);
    _Float16* h3  = take(Bsz * H3);
    _Float16* h4  = take(Bsz * H4);
    (void)ws_size;

    auto cvt = [&](const float* src, _Float16* dst, long n) {
        long thr = n / 8;
        cvt_f32_f16_kernel<<<(int)((thr + 255) / 256), 256, 0, stream>>>(src, dst, n);
    };
    cvt(x,  xh,  Bsz * DIN);
    cvt(w1, w1h, H1 * DIN);
    cvt(w2, w2h, DOUT * H1);
    cvt(w3, w3h, H3 * DOUT);
    cvt(w4, w4h, H4 * H3);
    cvt(w5, w5h, DOUT * H4);

    dim3 blk(256);
    const int shmem = LDS_TOTAL;
    gemm_f16_kernel<0, 1><<<dim3(H1 / 128, Bsz / 256), blk, shmem, stream>>>(
        xh, w1h, b1, nullptr, h1, nullptr, (int)Bsz, (int)H1, (int)DIN);
    gemm_f16_kernel<1, 1><<<dim3(DOUT / 128, Bsz / 256), blk, shmem, stream>>>(
        h1, w2h, b2, cb, h2, nullptr, (int)Bsz, (int)DOUT, (int)H1);
    gemm_f16_kernel<0, 1><<<dim3(H3 / 128, Bsz / 256), blk, shmem, stream>>>(
        h2, w3h, b3, nullptr, h3, nullptr, (int)Bsz, (int)H3, (int)DOUT);
    gemm_f16_kernel<0, 1><<<dim3(H4 / 128, Bsz / 256), blk, shmem, stream>>>(
        h3, w4h, b4, nullptr, h4, nullptr, (int)Bsz, (int)H4, (int)H3);
    gemm_f16_kernel<0, 0><<<dim3(DOUT / 128, Bsz / 256), blk, shmem, stream>>>(
        h4, w5h, b5, nullptr, nullptr, out, (int)Bsz, (int)DOUT, (int)H4);
}
